// Aligner_11871289606223
// MI455X (gfx1250) — matrix operand order repaired and checked by reading, not compiled, not run on hardware
//
#include <hip/hip_runtime.h>
#include <hip/hip_bf16.h>
#include <stdint.h>

// ---------- problem constants ----------
#define BB   4
#define LQ   1024
#define LK   4096
#define HID  384        // = H*DH = 2*192
#define KSTEPS (HID/32) // 12 k-steps of 32 for bf16 WMMA

typedef __bf16 bf16_t;
typedef __attribute__((ext_vector_type(16))) __bf16  v16bf;
typedef __attribute__((ext_vector_type(8)))  float   v8f;
typedef __attribute__((ext_vector_type(4)))  unsigned int v4u;

union BFrag { v16bf v; v4u u[2]; };

// round-to-nearest-even f32 -> bf16 (avoids relying on __bf16 conversions)
__device__ __forceinline__ bf16_t f2bf(float f) {
    unsigned int u = __float_as_uint(f);
    unsigned int r = (u + 0x7FFFu + ((u >> 16) & 1u)) >> 16;
    unsigned short s = (unsigned short)r;
    return __builtin_bit_cast(bf16_t, s);
}

// 16 bf16 fragment from two 16B-aligned groups of 8
__device__ __forceinline__ v16bf load_frag(const bf16_t* p0, const bf16_t* p1) {
    BFrag f;
    f.u[0] = *(const v4u*)p0;
    f.u[1] = *(const v4u*)p1;
    return f.v;
}

__device__ __forceinline__ v8f wmma_bf16(v16bf a, v16bf b, v8f c) {
    return __builtin_amdgcn_wmma_f32_16x16x32_bf16(
        /*neg_a=*/false, a, /*neg_b=*/false, b,
        /*c_mod=*/(short)0, c, /*reuse_a=*/false, /*reuse_b=*/false);
}

__device__ __forceinline__ v8f zero8() {
    v8f z = {0.f,0.f,0.f,0.f,0.f,0.f,0.f,0.f};
    return z;
}

// ---- CDNA5 async global->LDS (ASYNCcnt path), 16B per lane ----
__device__ __forceinline__ void async_b128(unsigned lds_addr, const void* gaddr) {
    asm volatile("global_load_async_to_lds_b128 %0, %1, off"
                 :: "v"(lds_addr), "v"(gaddr) : "memory");
}
__device__ __forceinline__ void wait_async0() {
    asm volatile("s_wait_asynccnt 0x0" ::: "memory");
}
// low 32 bits of a generic LDS pointer == wave-relative LDS byte address
__device__ __forceinline__ unsigned lds_addr_of(const void* p) {
    return (unsigned)(uintptr_t)p;
}

// ---------- fp32 -> bf16 elementwise ----------
__global__ void cvt_f32_bf16(const float* __restrict__ in, bf16_t* __restrict__ out, int n) {
    int i = blockIdx.x * blockDim.x + threadIdx.x;
    if (i < n) out[i] = f2bf(in[i]);
}

// ---------- W[k][n] (384x384 fp32) -> WT[n][k] bf16 ----------
__global__ void transpose_w(const float* __restrict__ W, bf16_t* __restrict__ WT) {
    int idx = blockIdx.x * blockDim.x + threadIdx.x;
    if (idx < HID * HID) {
        int n = idx / HID, k = idx % HID;
        WT[idx] = f2bf(W[(size_t)k * HID + n]);
    }
}

// ---------- projection GEMM: Y[M][384] = X[M][384] @ WT^T + bias ----------
// 8 waves / WG: 4 row-groups x 2 col-groups; wave = 16 rows x 192 cols.
// transpose_out: store Y as Vt[b][col][k] with k = row % 4096 (LK).
__launch_bounds__(256, 1)
__global__ void proj_gemm(const bf16_t* __restrict__ X,
                          const bf16_t* __restrict__ WT,   // [384][384], N-major
                          const float*  __restrict__ bias, // [384]
                          bf16_t* __restrict__ Y,
                          int transpose_out) {
    const int tid  = threadIdx.x;
    const int wave = tid >> 5;
    const int lane = tid & 31;
    const int l16  = lane & 15;
    const bool hi  = lane >= 16;

    const int r0 = blockIdx.x * 64 + (wave & 3) * 16;
    const int c0 = (wave >> 2) * 192;

    v8f acc[12];
#pragma unroll
    for (int f = 0; f < 12; ++f) acc[f] = zero8();

    const bf16_t* xrow = X + (size_t)(r0 + l16) * HID + (hi ? 8 : 0);
#pragma unroll
    for (int kk = 0; kk < KSTEPS; ++kk) {
        v16bf a = load_frag(xrow + kk * 32, xrow + kk * 32 + 16);
#pragma unroll
        for (int f = 0; f < 12; ++f) {
            const bf16_t* wb = WT + (size_t)(c0 + f * 16 + l16) * HID + kk * 32 + (hi ? 8 : 0);
            v16bf bfr = load_frag(wb, wb + 16);
            acc[f] = wmma_bf16(a, bfr, acc[f]);
        }
    }

#pragma unroll
    for (int f = 0; f < 12; ++f) {
        const int col = c0 + f * 16 + l16;
        const float bb = bias[col];
#pragma unroll
        for (int i = 0; i < 8; ++i) {
            const int row = r0 + i + (hi ? 8 : 0);
            const float v = acc[f][i] + bb;
            if (!transpose_out) {
                Y[(size_t)row * HID + col] = f2bf(v);
            } else {
                const int b  = row >> 12;     // row / LK
                const int rk = row & (LK - 1);
                Y[((size_t)b * HID + col) * LK + rk] = f2bf(v);
            }
        }
    }
}

// ---------- fused attention: softmax(Q K^T / sqrt(192) + mask) V ----------
// grid = B*(LQ/64); 4 waves/WG, wave = 16 q-rows x 384 ctx cols.
// Double-buffered K/V^T tiles staged via async global->LDS.
#define KTILE 32
__launch_bounds__(128, 1)
__global__ void attn_kernel(const bf16_t* __restrict__ Qb,   // [B*LQ][384]
                            const bf16_t* __restrict__ Kb,   // [B*LK][384]
                            const bf16_t* __restrict__ Vt,   // [B][384][LK]
                            const int*    __restrict__ mask, // [B][LQ][LK]
                            float* __restrict__ out) {       // [B][LQ][384]
    __shared__ bf16_t sK[2][KTILE * HID];   // 2 x 24 KB: K tile [key][dim]
    __shared__ bf16_t sV[2][HID * KTILE];   // 2 x 24 KB: V^T tile [dim][key]
    __shared__ bf16_t sP[4][16 * KTILE];    // 4 KB: per-wave probs [row][key]

    const int tid  = threadIdx.x;
    const int wave = tid >> 5;
    const int lane = tid & 31;
    const int l16  = lane & 15;
    const bool hi  = lane >= 16;

    const int wg = blockIdx.x;
    const int b  = wg / (LQ / 64);
    const int q0 = (wg % (LQ / 64)) * 64 + wave * 16;

    const bf16_t* kglob = Kb + (size_t)b * LK * HID;
    const bf16_t* vglob = Vt + (size_t)b * HID * LK;
    const int*    mglob = mask + ((size_t)b * LQ + q0) * LK;

    // ---- async tile stage: K tile is 32*384 bf16 contiguous (1536 b128),
    //      V^T tile is 384 rows x 32 keys (4 b128 per row). 128 threads. ----
    auto stage_tiles = [&](int buf, int kb) {
        const unsigned kdst = lds_addr_of(&sK[buf][0]) + (unsigned)tid * 16u;
        const char*    ksrc = (const char*)(kglob + (size_t)kb * HID) + (size_t)tid * 16;
#pragma unroll
        for (int it = 0; it < 12; ++it)
            async_b128(kdst + (unsigned)it * 2048u, ksrc + (size_t)it * 2048);
#pragma unroll
        for (int it = 0; it < 12; ++it) {
            const int j = tid + it * 128;
            const int d = j >> 2, c = j & 3;
            async_b128(lds_addr_of(&sV[buf][d * KTILE + c * 8]),
                       vglob + (size_t)d * LK + kb + c * 8);
        }
    };

    // ---- Q fragments in registers (16 rows x 384 dims) ----
    stage_tiles(0, 0);  // kick off first tile before doing anything else
    v16bf qa[KSTEPS];
    {
        const bf16_t* qrow = Qb + ((size_t)b * LQ + q0 + l16) * HID + (hi ? 8 : 0);
#pragma unroll
        for (int kk = 0; kk < KSTEPS; ++kk)
            qa[kk] = load_frag(qrow + kk * 32, qrow + kk * 32 + 16);
    }

    v8f ctx[24];
#pragma unroll
    for (int f = 0; f < 24; ++f) ctx[f] = zero8();
    float rmax[8], rsum[8];
#pragma unroll
    for (int i = 0; i < 8; ++i) { rmax[i] = -3.0e38f; rsum[i] = 0.0f; }

    const float scale = 0.0721687836487032f; // 1/sqrt(192)

    wait_async0();
    __syncthreads();

    for (int kb = 0; kb < LK; kb += KTILE) {
        const int cur = (kb / KTILE) & 1;
        // prefetch next tile into the other buffer while we compute
        if (kb + KTILE < LK) stage_tiles(cur ^ 1, kb + KTILE);

        // S = Q * K^T : two 16-key C-fragments
        const bf16_t* sKc = sK[cur];
        const bf16_t* sVc = sV[cur];
        v8f s0 = zero8(), s1 = zero8();
#pragma unroll
        for (int kk = 0; kk < KSTEPS; ++kk) {
            const int kd = kk * 32 + (hi ? 8 : 0);
            const bf16_t* b0 = sKc + (size_t)l16 * HID + kd;
            const bf16_t* b1 = sKc + (size_t)(16 + l16) * HID + kd;
            s0 = wmma_bf16(qa[kk], load_frag(b0, b0 + 16), s0);
            s1 = wmma_bf16(qa[kk], load_frag(b1, b1 + 16), s1);
        }

        // mask + scale + online softmax (row m lives in 16-lane half of VGPR i)
        float corr[8];
#pragma unroll
        for (int i = 0; i < 8; ++i) {
            const int q = i + (hi ? 8 : 0);
            const int* mrow = mglob + (size_t)q * LK + kb;
            const int m0 = __builtin_nontemporal_load(mrow + l16);
            const int m1 = __builtin_nontemporal_load(mrow + 16 + l16);
            const float a0 = s0[i] * scale + (m0 ? 0.0f : -1e9f);
            const float a1 = s1[i] * scale + (m1 ? 0.0f : -1e9f);
            float mx = fmaxf(a0, a1);
#pragma unroll
            for (int off = 1; off < 16; off <<= 1)
                mx = fmaxf(mx, __shfl_xor(mx, off, 32));
            const float mnew = fmaxf(rmax[i], mx);
            const float p0 = __expf(a0 - mnew);
            const float p1 = __expf(a1 - mnew);
            float ps = p0 + p1;
#pragma unroll
            for (int off = 1; off < 16; off <<= 1)
                ps += __shfl_xor(ps, off, 32);
            const float c = __expf(rmax[i] - mnew);
            rsum[i] = rsum[i] * c + ps;
            rmax[i] = mnew;
            corr[i] = c;
            // C-layout probs -> LDS (per-wave private region)
            bf16_t* prow = &sP[wave][q * KTILE];
            prow[l16]      = f2bf(p0);
            prow[16 + l16] = f2bf(p1);
        }

        // rescale running context
#pragma unroll
        for (int f = 0; f < 24; ++f)
#pragma unroll
            for (int i = 0; i < 8; ++i) ctx[f][i] = ctx[f][i] * corr[i];

        // same-wave DS RAW: make prob stores visible before A-fragment reload
        asm volatile("s_wait_dscnt 0x0" ::: "memory");

        // P as bf16 A-fragment (16 x 32)
        const bf16_t* pr = &sP[wave][(size_t)l16 * KTILE + (hi ? 8 : 0)];
        const v16bf pa = load_frag(pr, pr + 16);

        // ctx += P @ V  (24 column fragments of 16)
#pragma unroll
        for (int f = 0; f < 24; ++f) {
            const bf16_t* vb = sVc + (size_t)(f * 16 + l16) * KTILE + (hi ? 8 : 0);
            ctx[f] = wmma_bf16(pa, load_frag(vb, vb + 16), ctx[f]);
        }

        // next tile's async loads must have landed before we flip buffers
        wait_async0();
        __syncthreads();
    }

    // finalize: divide by rowsum, store fp32 (non-temporal, write-once)
    float inv[8];
#pragma unroll
    for (int i = 0; i < 8; ++i) inv[i] = 1.0f / rsum[i];
    float* obase = out + ((size_t)b * LQ + q0) * HID;
#pragma unroll
    for (int f = 0; f < 24; ++f) {
#pragma unroll
        for (int i = 0; i < 8; ++i) {
            const int q = i + (hi ? 8 : 0);
            __builtin_nontemporal_store(ctx[f][i] * inv[i],
                                        obase + (size_t)q * HID + f * 16 + l16);
        }
    }
}

// ---------- host launcher ----------
static inline size_t align256(size_t x) { return (x + 255) & ~(size_t)255; }

extern "C" void kernel_launch(void* const* d_in, const int* in_sizes, int n_in,
                              void* d_out, int out_size, void* d_ws, size_t ws_size,
                              hipStream_t stream) {
    const float* hs   = (const float*)d_in[0]; // [B,LQ,HID]
    const float* rhs  = (const float*)d_in[1]; // [B,LK,HID]
    const int*   msk  = (const int*)  d_in[2]; // [B,LQ,LK]
    const float* Wq   = (const float*)d_in[3];
    const float* bq   = (const float*)d_in[4];
    const float* Wk   = (const float*)d_in[5];
    const float* bk   = (const float*)d_in[6];
    const float* Wv   = (const float*)d_in[7];
    const float* bv   = (const float*)d_in[8];
    float* out = (float*)d_out;

    const size_t n_hs  = (size_t)BB * LQ * HID;   // 1,572,864
    const size_t n_rhs = (size_t)BB * LK * HID;   // 6,291,456
    const size_t n_w   = (size_t)HID * HID;       //   147,456

    char* ws = (char*)d_ws;
    size_t off = 0;
    bf16_t* hsb = (bf16_t*)(ws + off); off = align256(off + n_hs  * 2);
    bf16_t* rhb = (bf16_t*)(ws + off); off = align256(off + n_rhs * 2);
    bf16_t* wqT = (bf16_t*)(ws + off); off = align256(off + n_w   * 2);
    bf16_t* wkT = (bf16_t*)(ws + off); off = align256(off + n_w   * 2);
    bf16_t* wvT = (bf16_t*)(ws + off); off = align256(off + n_w   * 2);
    bf16_t* Qb  = (bf16_t*)(ws + off); off = align256(off + n_hs  * 2);
    bf16_t* Kb  = (bf16_t*)(ws + off); off = align256(off + n_rhs * 2);
    bf16_t* Vt  = (bf16_t*)(ws + off); off = align256(off + n_rhs * 2);
    (void)ws_size; (void)n_in; (void)in_sizes; (void)out_size;

    // 1) convert activations to bf16
    cvt_f32_bf16<<<(int)((n_hs  + 255) / 256), 256, 0, stream>>>(hs,  hsb, (int)n_hs);
    cvt_f32_bf16<<<(int)((n_rhs + 255) / 256), 256, 0, stream>>>(rhs, rhb, (int)n_rhs);
    // 2) transpose weights to N-major bf16
    transpose_w<<<(int)((n_w + 255) / 256), 256, 0, stream>>>(Wq, wqT);
    transpose_w<<<(int)((n_w + 255) / 256), 256, 0, stream>>>(Wk, wkT);
    transpose_w<<<(int)((n_w + 255) / 256), 256, 0, stream>>>(Wv, wvT);
    // 3) projections (Q row-major, K row-major, V transposed per batch)
    proj_gemm<<<(BB * LQ) / 64, 256, 0, stream>>>(hsb, wqT, bq, Qb, 0);
    proj_gemm<<<(BB * LK) / 64, 256, 0, stream>>>(rhb, wkT, bk, Kb, 0);
    proj_gemm<<<(BB * LK) / 64, 256, 0, stream>>>(rhb, wvT, bv, Vt, 1);
    // 4) fused masked-softmax attention (double-buffered async LDS staging)
    attn_kernel<<<BB * (LQ / 64), 128, 0, stream>>>(Qb, Kb, Vt, msk, out);
}
